// MultiHeadAttention_40492951667420
// MI455X (gfx1250) — matrix operand order repaired and checked by reading, not compiled, not run on hardware
//
#include <hip/hip_runtime.h>
#include <hip/hip_bf16.h>

// ---------------------------------------------------------------------------
// Transformer-XL relative multi-head attention for gfx1250 (MI455X).
// bf16 WMMA (v_wmma_f32_16x16x32_bf16) for every GEMM, fp32 accumulation.
// Double-buffered LDS staging with async global->LDS copies when available.
// rel_shift folded into the softmax pass as a read offset.
// ---------------------------------------------------------------------------

#define BB 2
#define TT 1024
#define MM 1024
#define DD 1024
#define HH 16
#define DKK 64
#define DVV 64
#define LL (MM + TT)        // 2048
#define RR (2 * TT + MM)    // 3072
#define HDK (HH * DKK)      // 1024
#define HDV (HH * DVV)      // 1024

typedef __attribute__((ext_vector_type(16))) __bf16 v16bf;
typedef __attribute__((ext_vector_type(8)))  float  v8f;
typedef __attribute__((ext_vector_type(4)))  int    i4v;
typedef unsigned short us;

union FragBF { v16bf v; unsigned int w[8]; };

#if defined(__gfx1250__) && __has_builtin(__builtin_amdgcn_global_load_async_to_lds_b128)
#define HAVE_ASYNC 1
typedef __attribute__((address_space(1))) i4v gi4;   // global int4
typedef __attribute__((address_space(3))) i4v li4;   // LDS int4
#else
#define HAVE_ASYNC 0
#endif

__device__ __forceinline__ void copy16(const us* g, us* l) {
#if HAVE_ASYNC
    // CDNA5 ASYNCcnt-tracked DMA of 16B global -> LDS (per lane addresses).
    __builtin_amdgcn_global_load_async_to_lds_b128((gi4*)g, (li4*)l, 0, 0);
#else
    *(uint4*)l = *(const uint4*)g;
#endif
}

__device__ __forceinline__ void wait_async0() {
#if HAVE_ASYNC
#if __has_builtin(__builtin_amdgcn_s_wait_asynccnt)
    __builtin_amdgcn_s_wait_asynccnt(0);
#else
    asm volatile("s_wait_asynccnt 0x0" ::: "memory");
#endif
#endif
}

__device__ __forceinline__ us f2bf(float f) {
    unsigned u = __float_as_uint(f);
    unsigned r = u + 0x7FFFu + ((u >> 16) & 1u);   // round-to-nearest-even
    return (us)(r >> 16);
}

__device__ __forceinline__ v8f wmma_bf16(const FragBF& a, const FragBF& b, v8f c) {
    return __builtin_amdgcn_wmma_f32_16x16x32_bf16(false, a.v, false, b.v,
                                                   (short)0, c, false, false);
}

// ---------------------------------------------------------------------------
// Elementwise helpers
// ---------------------------------------------------------------------------
__global__ void conv_f32_bf16(const float* __restrict__ in,
                              us* __restrict__ out, long long n) {
    long long i = (long long)blockIdx.x * blockDim.x + threadIdx.x;
    if (i < n) out[i] = f2bf(in[i]);
}

// out[cols][rows] (bf16) = transpose(in[rows][cols] fp32)
__global__ void transpose_conv(const float* __restrict__ in,
                               us* __restrict__ out, int rows, int cols) {
    long long i = (long long)blockIdx.x * blockDim.x + threadIdx.x;
    long long total = (long long)rows * cols;
    if (i >= total) return;
    int r = (int)(i / cols);
    int c = (int)(i % cols);
    out[(long long)c * rows + r] = f2bf(in[i]);
}

// values = concat(memory, inputs) along L, converted to bf16. [B, L, D]
__global__ void concat_conv(const float* __restrict__ memory,
                            const float* __restrict__ inputs,
                            us* __restrict__ out) {
    long long i = (long long)blockIdx.x * blockDim.x + threadIdx.x;
    long long total = (long long)BB * LL * DD;
    if (i >= total) return;
    int b = (int)(i / ((long long)LL * DD));
    int rem = (int)(i % ((long long)LL * DD));
    int l = rem / DD;
    int d = rem % DD;
    float v = (l < MM) ? memory[(long long)b * MM * DD + (long long)l * DD + d]
                       : inputs[(long long)b * TT * DD + (long long)(l - MM) * DD + d];
    out[i] = f2bf(v);
}

// Qw = bf16(Q + r_w_bias), Qr = bf16(Q + r_r_bias); bias indexed by column.
__global__ void bias_conv(const float* __restrict__ Q,
                          const float* __restrict__ rwb,
                          const float* __restrict__ rrb,
                          us* __restrict__ Qw,
                          us* __restrict__ Qr, long long n) {
    long long i = (long long)blockIdx.x * blockDim.x + threadIdx.x;
    if (i >= n) return;
    int c = (int)(i % HDK);
    float q = Q[i];
    Qw[i] = f2bf(q + rwb[c]);
    Qr[i] = f2bf(q + rrb[c]);
}

// ---------------------------------------------------------------------------
// WMMA GEMM: C[M,N] f32 = A[M,K] bf16 * B^T, where B element (k,n) is read
// from Bp[n*ldb + k] (every operand is pre-laid-out so this holds).
// Batched over blockIdx.z with separate per-b / per-h pointer offsets.
// Block: 256 threads = 8 waves. Block tile 128x64, wave tile 32x32
// (2 A-frags x 2 B-frags -> 4 WMMA per K-step). K-step 32, double-buffered
// LDS staged with async b128 copies. Requires M%128==0, N%64==0, K%32==0.
// ---------------------------------------------------------------------------
__global__ __launch_bounds__(256)
void gemm_bt_wmma(const us* __restrict__ A,
                  const us* __restrict__ Bm,
                  float* __restrict__ C,
                  int Mdim, int Ndim, int Kdim,
                  int lda, int ldb, int ldc,
                  long long aOffB, long long aOffH,
                  long long bOffB, long long bOffH,
                  long long cOffB, long long cOffH,
                  int Hn) {
    // rows padded to 40 ushorts = 80B: multiple of 16B (async b128 alignment),
    // 20-bank lane stride -> conflict-free fragment reads.
    __shared__ us At[2][128][40];
    __shared__ us Bt[2][64][40];

    const int z  = blockIdx.z;
    const int bb = z / Hn;
    const int hh = z % Hn;
    const us* Ap = A  + (long long)bb * aOffB + (long long)hh * aOffH;
    const us* Bp = Bm + (long long)bb * bOffB + (long long)hh * bOffH;
    float*    Cp = C  + (long long)bb * cOffB + (long long)hh * cOffH;

    const int n0   = blockIdx.x * 64;
    const int m0   = blockIdx.y * 128;
    const int tid  = threadIdx.x;
    const int lane = tid & 31;
    const int w    = tid >> 5;
    const int wm   = w & 3;    // M sub-tile (32 rows each)
    const int wn   = w >> 2;   // N sub-tile (32 cols each)

    v8f acc00 = {}, acc01 = {}, acc10 = {}, acc11 = {};

    auto stage = [&](int buf, int k0) {
        // A tile: 128 rows x 32 K  (512 x 16B tasks, 2 per thread)
#pragma unroll
        for (int t = 0; t < 2; ++t) {
            int task = tid + t * 256;
            int r = task >> 2, ck = (task & 3) * 8;
            copy16(Ap + (long long)(m0 + r) * lda + k0 + ck, &At[buf][r][ck]);
        }
        // B tile: 64 n-rows x 32 K (256 x 16B tasks, 1 per thread)
        {
            int r = tid >> 2, ck = (tid & 3) * 8;
            copy16(Bp + (long long)(n0 + r) * ldb + k0 + ck, &Bt[buf][r][ck]);
        }
    };

    stage(0, 0);
    wait_async0();
    __syncthreads();

    const int nK = Kdim >> 5;
    for (int ki = 0; ki < nK; ++ki) {
        const int cur = ki & 1;
        if (ki + 1 < nK) stage(cur ^ 1, (ki + 1) << 5);   // overlap with compute

        const int mr = lane & 15, hl = lane >> 4;
        FragBF a0, a1, b0, b1;
#pragma unroll
        for (int j = 0; j < 8; ++j) {
            // A layout: lane half hl covers K = hl*8 + [0..7] and 16 + hl*8 + [0..7]
            int kk = (j < 4) ? (hl * 8 + 2 * j) : (hl * 8 + 2 * j + 8);
            a0.w[j] = *(const unsigned int*)&At[cur][wm * 32 + mr][kk];
            a1.w[j] = *(const unsigned int*)&At[cur][wm * 32 + 16 + mr][kk];
            // B layout: lane = N column, half hl covers K = hl*8 + [0..7] and 16 + hl*8 + [0..7]
            b0.w[j] = *(const unsigned int*)&Bt[cur][wn * 32 + mr][kk];
            b1.w[j] = *(const unsigned int*)&Bt[cur][wn * 32 + 16 + mr][kk];
        }
        acc00 = wmma_bf16(a0, b0, acc00);
        acc01 = wmma_bf16(a0, b1, acc01);
        acc10 = wmma_bf16(a1, b0, acc10);
        acc11 = wmma_bf16(a1, b1, acc11);

        wait_async0();          // next buffer landed
        __syncthreads();
    }

    // D layout: VGPR r -> row hl*8 + r, lane%16 -> column
    const int nn = lane & 15, hl = lane >> 4;
#pragma unroll
    for (int r = 0; r < 8; ++r) {
        float* d0 = Cp + (long long)(m0 + wm * 32 + hl * 8 + r) * ldc + n0 + wn * 32 + nn;
        d0[0]  = acc00[r];
        d0[16] = acc01[r];
        float* d1 = Cp + (long long)(m0 + wm * 32 + 16 + hl * 8 + r) * ldc + n0 + wn * 32 + nn;
        d1[0]  = acc10[r];
        d1[16] = acc11[r];
    }
}

// ---------------------------------------------------------------------------
// Fused rel-shift + mask + softmax. One block (256 thr) per (b,h,t) row.
// logits[l] = (content[l] + rel[l + T - t]) * scale - (1-mask)*1e30
// ---------------------------------------------------------------------------
__global__ __launch_bounds__(256)
void softmax_shift(const float* __restrict__ content,
                   const float* __restrict__ rel,
                   const float* __restrict__ mask,
                   us* __restrict__ probs) {
    const int rid = blockIdx.x;            // (b*H + h)*T + t
    const int t  = rid % TT;
    const int bh = rid / TT;
    const int b  = bh / HH;
    const float* crow = content + (long long)rid * LL;
    const float* rrow = rel + (long long)rid * RR + (TT - t);  // rel_shift
    const float* mrow = mask + ((long long)b * TT + t) * LL;
    us* prow = probs + (long long)rid * LL;

    const float scale = 0.125f;            // DK^-0.5, DK=64
    const int tid = threadIdx.x;
    float sv[LL / 256];
    float mx = -3.4e38f;
#pragma unroll
    for (int i = 0; i < LL / 256; ++i) {
        int l = tid + i * 256;
        float s = (crow[l] + rrow[l]) * scale - (1.0f - mrow[l]) * 1e30f;
        sv[i] = s;
        mx = fmaxf(mx, s);
    }
    __shared__ float red[256];
    red[tid] = mx; __syncthreads();
    for (int off = 128; off > 0; off >>= 1) {
        if (tid < off) red[tid] = fmaxf(red[tid], red[tid + off]);
        __syncthreads();
    }
    mx = red[0]; __syncthreads();

    float sum = 0.0f;
#pragma unroll
    for (int i = 0; i < LL / 256; ++i) {
        float e = __expf(sv[i] - mx);
        sv[i] = e;
        sum += e;
    }
    red[tid] = sum; __syncthreads();
    for (int off = 128; off > 0; off >>= 1) {
        if (tid < off) red[tid] += red[tid + off];
        __syncthreads();
    }
    float inv = 1.0f / red[0];
#pragma unroll
    for (int i = 0; i < LL / 256; ++i)
        prow[tid + i * 256] = f2bf(sv[i] * inv);
}

// ---------------------------------------------------------------------------
// Host-side orchestration
// ---------------------------------------------------------------------------
extern "C" void kernel_launch(void* const* d_in, const int* in_sizes, int n_in,
                              void* d_out, int out_size, void* d_ws, size_t ws_size,
                              hipStream_t stream) {
    const float* inputs = (const float*)d_in[0];   // [B,T,D]
    const float* amask  = (const float*)d_in[1];   // [B,T,L]
    const float* posenc = (const float*)d_in[2];   // [R,D]
    const float* memory = (const float*)d_in[3];   // [B,M,D]
    const float* Wq     = (const float*)d_in[4];   // [D, H*DK]
    const float* Wk     = (const float*)d_in[5];
    const float* Wv     = (const float*)d_in[6];
    const float* Wr     = (const float*)d_in[7];
    const float* rwb    = (const float*)d_in[8];   // [H*DK]
    const float* rrb    = (const float*)d_in[9];
    const float* Wout   = (const float*)d_in[10];  // [H*DV, D]
    float* out = (float*)d_out;                    // [B,T,D] fp32

    // bump allocator over d_ws
    char* base = (char*)d_ws;
    size_t off = 0;
    auto alloc = [&](size_t bytes) -> void* {
        void* p = base + off;
        off = (off + bytes + 255) & ~(size_t)255;
        return p;
    };
    us* Xb    = (us*)alloc((size_t)BB * TT * DD * 2);
    us* Vb    = (us*)alloc((size_t)BB * LL * DD * 2);
    us* Pb    = (us*)alloc((size_t)RR * DD * 2);
    us* Wqbt  = (us*)alloc((size_t)HDK * DD * 2);   // [HDK, D] = Wq^T
    us* Wkbt  = (us*)alloc((size_t)HDK * DD * 2);
    us* Wvbt  = (us*)alloc((size_t)HDV * DD * 2);
    us* Wrbt  = (us*)alloc((size_t)HDK * DD * 2);
    us* Wobt  = (us*)alloc((size_t)DD * HDV * 2);   // [D, HDV] = Wout^T
    float* Qf  = (float*)alloc((size_t)BB * TT * HDK * 4);
    float* Kf  = (float*)alloc((size_t)BB * LL * HDK * 4);
    float* Vf  = (float*)alloc((size_t)BB * LL * HDV * 4);
    float* Rkf = (float*)alloc((size_t)RR * HDK * 4);
    us* Qwb  = (us*)alloc((size_t)BB * TT * HDK * 2);
    us* Qrb  = (us*)alloc((size_t)BB * TT * HDK * 2);
    us* Kb   = (us*)alloc((size_t)BB * LL * HDK * 2);
    us* VvbT = (us*)alloc((size_t)BB * HDV * LL * 2);  // [B, H*DV, L] = Vv^T per b
    us* Rkb  = (us*)alloc((size_t)RR * HDK * 2);
    float* contentf = (float*)alloc((size_t)BB * HH * TT * LL * 4);
    float* relf     = (float*)alloc((size_t)BB * HH * TT * RR * 4);
    us* probs = (us*)alloc((size_t)BB * HH * TT * LL * 2);
    float* attnf = (float*)alloc((size_t)BB * TT * HDV * 4);
    us* attnb = (us*)alloc((size_t)BB * TT * HDV * 2);

    const int TPB = 256;
    auto g1 = [&](long long n) { return dim3((unsigned)((n + TPB - 1) / TPB)); };

    // 1) bf16 conversion / layout passes
    conv_f32_bf16<<<g1((long long)BB * TT * DD), TPB, 0, stream>>>(inputs, Xb, (long long)BB * TT * DD);
    concat_conv<<<g1((long long)BB * LL * DD), TPB, 0, stream>>>(memory, inputs, Vb);
    conv_f32_bf16<<<g1((long long)RR * DD), TPB, 0, stream>>>(posenc, Pb, (long long)RR * DD);
    transpose_conv<<<g1((long long)DD * HDK), TPB, 0, stream>>>(Wq, Wqbt, DD, HDK);
    transpose_conv<<<g1((long long)DD * HDK), TPB, 0, stream>>>(Wk, Wkbt, DD, HDK);
    transpose_conv<<<g1((long long)DD * HDV), TPB, 0, stream>>>(Wv, Wvbt, DD, HDV);
    transpose_conv<<<g1((long long)DD * HDK), TPB, 0, stream>>>(Wr, Wrbt, DD, HDK);
    transpose_conv<<<g1((long long)HDV * DD), TPB, 0, stream>>>(Wout, Wobt, HDV, DD);

    // 2) projections (single batch, Hn=1); weights pre-transposed -> BT access
    gemm_bt_wmma<<<dim3(HDK / 64, (BB * TT) / 128, 1), dim3(256), 0, stream>>>(
        Xb, Wqbt, Qf, BB * TT, HDK, DD, DD, DD, HDK, 0, 0, 0, 0, 0, 0, 1);
    gemm_bt_wmma<<<dim3(HDK / 64, (BB * LL) / 128, 1), dim3(256), 0, stream>>>(
        Vb, Wkbt, Kf, BB * LL, HDK, DD, DD, DD, HDK, 0, 0, 0, 0, 0, 0, 1);
    gemm_bt_wmma<<<dim3(HDV / 64, (BB * LL) / 128, 1), dim3(256), 0, stream>>>(
        Vb, Wvbt, Vf, BB * LL, HDV, DD, DD, DD, HDV, 0, 0, 0, 0, 0, 0, 1);
    gemm_bt_wmma<<<dim3(HDK / 64, RR / 128, 1), dim3(256), 0, stream>>>(
        Pb, Wrbt, Rkf, RR, HDK, DD, DD, DD, HDK, 0, 0, 0, 0, 0, 0, 1);

    // 3) biases + bf16 repacks (K natural [L,DK] layout is already B^T)
    bias_conv<<<g1((long long)BB * TT * HDK), TPB, 0, stream>>>(
        Qf, rwb, rrb, Qwb, Qrb, (long long)BB * TT * HDK);
    conv_f32_bf16<<<g1((long long)BB * LL * HDK), TPB, 0, stream>>>(Kf, Kb, (long long)BB * LL * HDK);
    conv_f32_bf16<<<g1((long long)RR * HDK), TPB, 0, stream>>>(Rkf, Rkb, (long long)RR * HDK);
    for (int b = 0; b < BB; ++b)
        transpose_conv<<<g1((long long)LL * HDV), TPB, 0, stream>>>(
            Vf + (long long)b * LL * HDV, VvbT + (long long)b * HDV * LL, LL, HDV);

    // 4) logits GEMMs (batched over z = b*H + h)
    gemm_bt_wmma<<<dim3(LL / 64, TT / 128, BB * HH), dim3(256), 0, stream>>>(
        Qwb, Kb, contentf, TT, LL, DKK,
        HDK, HDK, LL,
        (long long)TT * HDK, DKK,
        (long long)LL * HDK, DKK,
        (long long)HH * TT * LL, (long long)TT * LL, HH);
    gemm_bt_wmma<<<dim3(RR / 64, TT / 128, BB * HH), dim3(256), 0, stream>>>(
        Qrb, Rkb, relf, TT, RR, DKK,
        HDK, HDK, RR,
        (long long)TT * HDK, DKK,
        0, DKK,
        (long long)HH * TT * RR, (long long)TT * RR, HH);

    // 5) fused rel-shift + mask + softmax -> bf16 probs
    softmax_shift<<<dim3(BB * HH * TT), TPB, 0, stream>>>(contentf, relf, amask, probs);

    // 6) attn = probs @ Vv, using pre-transposed VvbT [B, H*DV, L]
    gemm_bt_wmma<<<dim3(DVV / 64, TT / 128, BB * HH), dim3(256), 0, stream>>>(
        probs, VvbT, attnf, TT, DVV, LL,
        LL, LL, HDV,
        (long long)HH * TT * LL, (long long)TT * LL,
        (long long)HDV * LL, (long long)DVV * LL,
        (long long)TT * HDV, DVV, HH);

    // 7) out = attn @ W_out (pre-transposed) : [B*T, D] fp32 -> d_out
    conv_f32_bf16<<<g1((long long)BB * TT * HDV), TPB, 0, stream>>>(attnf, attnb, (long long)BB * TT * HDV);
    gemm_bt_wmma<<<dim3(DD / 64, (BB * TT) / 128, 1), dim3(256), 0, stream>>>(
        attnb, Wobt, out, BB * TT, DD, HDV, HDV, HDV, DD, 0, 0, 0, 0, 0, 0, 1);
}
